// RPEAttention_9869834846446
// MI455X (gfx1250) — matrix operand order repaired and checked by reading, not compiled, not run on hardware
//
#include <hip/hip_runtime.h>

// ---------------------------------------------------------------------------
// iRPE attention (euc/ctx/shared-head/skip=1, rpe on k) for MI455X (gfx1250).
// bf16 WMMA (v_wmma_f32_16x16x32_bf16) everywhere, flash-style fused softmax,
// async global->LDS staging (ASYNCcnt) with double buffering.
// ---------------------------------------------------------------------------

typedef __attribute__((ext_vector_type(2)))  __bf16 v2bf;
typedef __attribute__((ext_vector_type(16))) __bf16 v16bf;
typedef __attribute__((ext_vector_type(8)))  float  v8f;
typedef int v4i_ __attribute__((vector_size(16)));   // b128 payload type

#define BATCH   8
#define H_HEADS 12
#define HD      64
#define N_SEQ   1025
#define NP      1152              // padded sequence rows (9 * 128)
#define C_DIM   768
#define ROWS_T  (BATCH * N_SEQ)   // 8200
#define RPAD    8320              // 65 * 128, padded GEMM M extent
#define NJ      1040              // 65*16, j extent covered by the flash loop

__device__ inline v8f wmma_bf16(v16bf a, v16bf b, v8f c) {
  return __builtin_amdgcn_wmma_f32_16x16x32_bf16(false, a, false, b, (short)0, c,
                                                 false, false);
}

// Async 16-byte global -> LDS copy (GLOBAL_LOAD_ASYNC_TO_LDS_B128, ASYNCcnt).
__device__ inline void async_cp16(const __bf16* g, __bf16* l) {
#if __has_builtin(__builtin_amdgcn_global_load_async_to_lds_b128)
  __builtin_amdgcn_global_load_async_to_lds_b128(
      (__attribute__((address_space(1))) v4i_*)(v4i_*)(void*)(__bf16*)g,
      (__attribute__((address_space(3))) v4i_*)(v4i_*)(void*)l, 0, 0);
#else
  unsigned lo = (unsigned)(unsigned long long)
      (__attribute__((address_space(3))) void*)(void*)l;
  asm volatile("global_load_async_to_lds_b128 %0, %1, off"
               :: "v"(lo), "v"((unsigned long long)g) : "memory");
#endif
}

#define WAIT_ASYNC() asm volatile("s_wait_asynccnt 0x0" ::: "memory")
#define WAIT_DS()    asm volatile("s_wait_dscnt 0x0" ::: "memory")

// A-matrix 16x32 bf16 fragment from row-major storage (rows 0..15 of p).
// lanes 0-15 hold M=lane, K 0..15; lanes 16-31 hold M=lane-16, K 16..31
// (contiguous order, matching the B loaders).
__device__ inline v16bf load_A(const __bf16* p, int stride, int lane, int kofs) {
  int row = lane & 15;
  int kb  = ((lane >> 4) << 4) + kofs;
  v16bf a;
#pragma unroll
  for (int v = 0; v < 8; ++v) {
    int k = (v << 1) + kb;
    v2bf pr = *(const v2bf*)(p + row * stride + k);
    a[2 * v]     = pr[0];
    a[2 * v + 1] = pr[1];
  }
  return a;
}

// B-matrix 32x16 fragment where element [k][n] = p[n*stride + kofs + k].
__device__ inline v16bf load_B_nk(const __bf16* p, int stride, int lane, int kofs) {
  int n  = lane & 15;
  int kb = ((lane >> 4) << 4) + kofs;
  v16bf b;
#pragma unroll
  for (int v = 0; v < 8; ++v) {
    v2bf pr = *(const v2bf*)(p + n * stride + kb + 2 * v);
    b[2 * v]     = pr[0];
    b[2 * v + 1] = pr[1];
  }
  return b;
}

// B-matrix 32x16 fragment where element [k][n] = p[k*stride + n] (k-major).
__device__ inline v16bf load_B_kn(const __bf16* p, int stride, int lane) {
  int n  = lane & 15;
  int kb = (lane >> 4) << 4;
  v16bf b;
#pragma unroll
  for (int v = 0; v < 8; ++v) {
    b[2 * v]     = p[(kb + 2 * v) * stride + n];
    b[2 * v + 1] = p[(kb + 2 * v + 1) * stride + n];
  }
  return b;
}

// ---------------------------------------------------------------------------
__global__ __launch_bounds__(256) void cvt_bf16(const float* __restrict__ src,
                                                __bf16* __restrict__ dst, long n) {
  long i = (long)blockIdx.x * blockDim.x + threadIdx.x;
  long st = (long)gridDim.x * blockDim.x;
  for (; i < n; i += st) dst[i] = (__bf16)src[i];
}

__global__ __launch_bounds__(256) void zero_bf16(__bf16* __restrict__ dst, long n) {
  long i = (long)blockIdx.x * blockDim.x + threadIdx.x;
  long st = (long)gridDim.x * blockDim.x;
  for (; i < n; i += st) dst[i] = (__bf16)0.0f;
}

// ---------------------------------------------------------------------------
// QKV GEMM: (RPAD x 768) @ (2304 x 768)^T, scattered into per-head q/k/v
// buffers (B*H, NP, 64) bf16; q pre-scaled by hd^-0.5 = 0.125.
// 128(M) x 64(N) block tile, K-step 32, async double-buffered LDS staging.
#define MT  128
#define NT  64
#define KT  32
#define LDT 40   // LDS row stride (80B: 16B aligned, conflict-free)

__global__ __launch_bounds__(256) void qkv_gemm(const __bf16* __restrict__ X,
                                                const __bf16* __restrict__ W,
                                                __bf16* __restrict__ qb,
                                                __bf16* __restrict__ kb,
                                                __bf16* __restrict__ vb) {
  __shared__ __bf16 xs[2][MT * LDT];
  __shared__ __bf16 wt[2][NT * LDT];
  int tid = threadIdx.x, lane = tid & 31, wave = tid >> 5;
  int rBase = blockIdx.x * MT;
  int cBase = blockIdx.y * NT;
  int wm = wave & 3, wn = wave >> 2;
  int sr = tid >> 2, sc = (tid & 3) * 8;   // staging row / 16B chunk

  v8f acc[2][2];
#pragma unroll
  for (int mi = 0; mi < 2; ++mi)
#pragma unroll
    for (int ni = 0; ni < 2; ++ni)
      acc[mi][ni] = (v8f){0.f, 0.f, 0.f, 0.f, 0.f, 0.f, 0.f, 0.f};

  // prologue: stage k0 = 0 into buffer 0
  async_cp16(X + (long)(rBase + sr) * C_DIM + sc, &xs[0][sr * LDT + sc]);
  async_cp16(X + (long)(rBase + sr + 64) * C_DIM + sc, &xs[0][(sr + 64) * LDT + sc]);
  async_cp16(W + (long)(cBase + sr) * C_DIM + sc, &wt[0][sr * LDT + sc]);

  for (int k0 = 0; k0 < C_DIM; k0 += KT) {
    int cur = (k0 >> 5) & 1;
    WAIT_ASYNC();
    __syncthreads();
    if (k0 + KT < C_DIM) {
      int kn = k0 + KT, nxt = cur ^ 1;
      async_cp16(X + (long)(rBase + sr) * C_DIM + kn + sc,
                 &xs[nxt][sr * LDT + sc]);
      async_cp16(X + (long)(rBase + sr + 64) * C_DIM + kn + sc,
                 &xs[nxt][(sr + 64) * LDT + sc]);
      async_cp16(W + (long)(cBase + sr) * C_DIM + kn + sc,
                 &wt[nxt][sr * LDT + sc]);
    }
    const __bf16* xa = &xs[cur][(wm * 32) * LDT];
    const __bf16* wb = &wt[cur][(wn * 32) * LDT];
    v16bf A0 = load_A(xa, LDT, lane, 0);
    v16bf A1 = load_A(xa + 16 * LDT, LDT, lane, 0);
    v16bf B0 = load_B_nk(wb, LDT, lane, 0);
    v16bf B1 = load_B_nk(wb + 16 * LDT, LDT, lane, 0);
    acc[0][0] = wmma_bf16(A0, B0, acc[0][0]);
    acc[0][1] = wmma_bf16(A0, B1, acc[0][1]);
    acc[1][0] = wmma_bf16(A1, B0, acc[1][0]);
    acc[1][1] = wmma_bf16(A1, B1, acc[1][1]);
  }

  int half = lane >> 4, nn = lane & 15;
#pragma unroll
  for (int mi = 0; mi < 2; ++mi)
#pragma unroll
    for (int ni = 0; ni < 2; ++ni) {
      int col = cBase + wn * 32 + ni * 16;     // 16-col tile, uniform t/h
      int t = col / C_DIM;
      int rem = col - t * C_DIM;
      int hh = rem >> 6;
      int d = (rem & 63) + nn;
#pragma unroll
      for (int vv = 0; vv < 8; ++vv) {
        int rr = rBase + wm * 32 + mi * 16 + vv + (half << 3);
        if (rr < ROWS_T) {
          int bI = rr / N_SEQ, nI = rr - bI * N_SEQ;
          long idx = (((long)(bI * H_HEADS + hh)) * NP + nI) * HD + d;
          float val = acc[mi][ni][vv];
          if (t == 0)      qb[idx] = (__bf16)(val * 0.125f);
          else if (t == 1) kb[idx] = (__bf16)val;
          else             vb[idx] = (__bf16)val;
        }
      }
    }
}

// ---------------------------------------------------------------------------
// Fused flash attention with contextual RPE bias on keys.
// grid = (B*H, 9); block = 256 (8 waves, all share one (b,h) and j-sequence);
// K/V tiles (32 x 64) async-staged into LDS, double-buffered.
#define LKV 72   // K/V LDS row stride (144B: 16B aligned, conflict-free)

__global__ __launch_bounds__(256) void flash_rpe(const __bf16* __restrict__ qb,
                                                 const __bf16* __restrict__ kb,
                                                 const __bf16* __restrict__ vb,
                                                 const float*  __restrict__ rpe_w,
                                                 __bf16* __restrict__ obf) {
  __shared__ __bf16 rpeLds[64 * 16];         // rpe_w padded 8 -> 16 buckets
  __shared__ unsigned char lutLds[63 * 63];  // euclid bucket LUT over (dr,dc)
  __shared__ float lookLds[8][16][16];       // per-wave lookup tile (q @ rpe)
  __shared__ __bf16 pshLds[8][16 * 32];      // per-wave P staging for PV wmma
  __shared__ __bf16 ksh[2][32 * LKV];        // double-buffered K tile
  __shared__ __bf16 vsh[2][32 * LKV];        // double-buffered V tile

  int tid = threadIdx.x, lane = tid & 31, wave = tid >> 5;
  int bh = blockIdx.x;
  long bhbase = (long)bh * NP;

  // staging assignment: 256 threads x 1 async b128 each per tile pair
  int smat = tid >> 7;             // 0 = K, 1 = V
  int srow = (tid >> 2) & 31;
  int scol = (tid & 3) * 8;
  const __bf16* skv = (smat ? vb : kb) + (bhbase + srow) * HD + scol;

  for (int i = tid; i < 64 * 16; i += 256) {
    int d = i >> 4, m = i & 15;
    rpeLds[i] = (__bf16)(m < 8 ? rpe_w[d * 8 + m] : 0.0f);
  }
  for (int i = tid; i < 63 * 63; i += 256) {
    int dr = i / 63 - 31, dc = i % 63 - 31;
    float dis = roundf(sqrtf((float)(dr * dr + dc * dc)));
    int bkt;
    if (dis <= 1.9f) {
      bkt = (int)roundf(dis);
    } else {                                  // alpha=1.9 beta=3.8 gamma=15.2
      float li = roundf(1.9f + logf(dis / 1.9f) * (1.9f / logf(8.0f)));
      li = fminf(li, 3.8f);
      bkt = (int)li;
    }
    lutLds[i] = (unsigned char)bkt;
  }

  // prologue: stage j = 0 into buffer 0 (rows j..j+31 of K and V)
  {
    __bf16* dst = (smat ? &vsh[0][0] : &ksh[0][0]) + srow * LKV + scol;
    async_cp16(skv, dst);
  }
  __syncthreads();

  int bI = bh / H_HEADS, hh = bh - bI * H_HEADS;
  int rowBase = blockIdx.y * 128 + wave * 16;   // < NP by construction
  long qoff = (bhbase + rowBase) * HD;
  v16bf Q0 = load_A(qb + qoff, HD, lane, 0);    // q already scaled by 0.125
  v16bf Q1 = load_A(qb + qoff, HD, lane, 32);

  // lookup[i, m] = q[i,:] @ rpe_w[:, m]  (cols 8..15 are zero padding)
  v8f L = (v8f){0.f, 0.f, 0.f, 0.f, 0.f, 0.f, 0.f, 0.f};
  L = wmma_bf16(Q0, load_B_kn(rpeLds, 16, lane), L);
  L = wmma_bf16(Q1, load_B_kn(rpeLds + 32 * 16, 16, lane), L);
  int half = lane >> 4, nn = lane & 15;
#pragma unroll
  for (int vv = 0; vv < 8; ++vv) lookLds[wave][vv + (half << 3)][nn] = L[vv];
  WAIT_DS();

  float mrow[8], lrow[8];
  v8f O[4];
#pragma unroll
  for (int vv = 0; vv < 8; ++vv) { mrow[vv] = -1e30f; lrow[vv] = 0.f; }
#pragma unroll
  for (int t = 0; t < 4; ++t) O[t] = (v8f){0.f, 0.f, 0.f, 0.f, 0.f, 0.f, 0.f, 0.f};

  for (int j = 0; j < NJ; j += 32) {
    int cur = (j >> 5) & 1;
    WAIT_ASYNC();
    __syncthreads();                 // tile j ready block-wide; prev reads done
    if (j + 32 < NJ) {               // stage tile j+32 into the other buffer
      __bf16* dst = (smat ? &vsh[cur ^ 1][0] : &ksh[cur ^ 1][0]) + srow * LKV + scol;
      async_cp16(skv + (long)(j + 32) * HD, dst);
    }

    const __bf16* kp = &ksh[cur][0];
    v8f S0 = (v8f){0.f, 0.f, 0.f, 0.f, 0.f, 0.f, 0.f, 0.f};
    v8f S1 = S0;
    S0 = wmma_bf16(Q0, load_B_nk(kp, LKV, lane, 0), S0);
    S0 = wmma_bf16(Q1, load_B_nk(kp, LKV, lane, 32), S0);
    S1 = wmma_bf16(Q0, load_B_nk(kp + 16 * LKV, LKV, lane, 0), S1);
    S1 = wmma_bf16(Q1, load_B_nk(kp + 16 * LKV, LKV, lane, 32), S1);

#pragma unroll
    for (int vv = 0; vv < 8; ++vv) {
      int m = vv + (half << 3);
      int gi = rowBase + m;
      float s0 = S0[vv], s1 = S1[vv];
      int gj0 = j + nn, gj1 = j + 16 + nn;
      if (gi < N_SEQ && gj0 < N_SEQ) {
        int bkt = 7;
        if (gi != 0 && gj0 != 0) {
          int pi = gi - 1, pj = gj0 - 1;
          bkt = lutLds[(((pi >> 5) - (pj >> 5)) + 31) * 63 +
                       ((pi & 31) - (pj & 31)) + 31];
        }
        s0 += lookLds[wave][m][bkt];
      } else if (gj0 >= N_SEQ) s0 = -1e30f;
      if (gi < N_SEQ && gj1 < N_SEQ) {
        int bkt = 7;
        if (gi != 0 && gj1 != 0) {
          int pi = gi - 1, pj = gj1 - 1;
          bkt = lutLds[(((pi >> 5) - (pj >> 5)) + 31) * 63 +
                       ((pi & 31) - (pj & 31)) + 31];
        }
        s1 += lookLds[wave][m][bkt];
      } else if (gj1 >= N_SEQ) s1 = -1e30f;

      // online softmax: 16-lane row reductions (wave32, xor masks stay in half)
      float tmax = fmaxf(s0, s1);
#pragma unroll
      for (int msk = 1; msk < 16; msk <<= 1)
        tmax = fmaxf(tmax, __shfl_xor(tmax, msk, 32));
      float mnew = fmaxf(mrow[vv], tmax);
      float alpha = __expf(mrow[vv] - mnew);
      float p0 = __expf(s0 - mnew);
      float p1 = __expf(s1 - mnew);
      float ps = p0 + p1;
#pragma unroll
      for (int msk = 1; msk < 16; msk <<= 1) ps += __shfl_xor(ps, msk, 32);
      lrow[vv] = lrow[vv] * alpha + ps;
      mrow[vv] = mnew;
      O[0][vv] *= alpha; O[1][vv] *= alpha; O[2][vv] *= alpha; O[3][vv] *= alpha;
      pshLds[wave][m * 32 + nn]      = (__bf16)p0;
      pshLds[wave][m * 32 + 16 + nn] = (__bf16)p1;
    }
    WAIT_DS();

    v16bf P = load_A(&pshLds[wave][0], 32, lane, 0);
    const __bf16* vp = &vsh[cur][0];
    O[0] = wmma_bf16(P, load_B_kn(vp + 0, LKV, lane), O[0]);
    O[1] = wmma_bf16(P, load_B_kn(vp + 16, LKV, lane), O[1]);
    O[2] = wmma_bf16(P, load_B_kn(vp + 32, LKV, lane), O[2]);
    O[3] = wmma_bf16(P, load_B_kn(vp + 48, LKV, lane), O[3]);
  }

#pragma unroll
  for (int vv = 0; vv < 8; ++vv) {
    int gi = rowBase + vv + (half << 3);
    if (gi < N_SEQ) {
      float inv = 1.0f / lrow[vv];
      long base = ((long)(bI * N_SEQ + gi)) * C_DIM + hh * HD;
#pragma unroll
      for (int t = 0; t < 4; ++t)
        obf[base + t * 16 + nn] = (__bf16)(O[t][vv] * inv);
    }
  }
}

// ---------------------------------------------------------------------------
// Output projection: (RPAD x 768) @ (768 x 768)^T + bias -> f32.
__global__ __launch_bounds__(256) void proj_gemm(const __bf16* __restrict__ X,
                                                 const __bf16* __restrict__ W,
                                                 const float* __restrict__ Pb,
                                                 float* __restrict__ out) {
  __shared__ __bf16 xs[2][MT * LDT];
  __shared__ __bf16 wt[2][NT * LDT];
  int tid = threadIdx.x, lane = tid & 31, wave = tid >> 5;
  int rBase = blockIdx.x * MT;
  int cBase = blockIdx.y * NT;
  int wm = wave & 3, wn = wave >> 2;
  int sr = tid >> 2, sc = (tid & 3) * 8;

  v8f acc[2][2];
#pragma unroll
  for (int mi = 0; mi < 2; ++mi)
#pragma unroll
    for (int ni = 0; ni < 2; ++ni)
      acc[mi][ni] = (v8f){0.f, 0.f, 0.f, 0.f, 0.f, 0.f, 0.f, 0.f};

  async_cp16(X + (long)(rBase + sr) * C_DIM + sc, &xs[0][sr * LDT + sc]);
  async_cp16(X + (long)(rBase + sr + 64) * C_DIM + sc, &xs[0][(sr + 64) * LDT + sc]);
  async_cp16(W + (long)(cBase + sr) * C_DIM + sc, &wt[0][sr * LDT + sc]);

  for (int k0 = 0; k0 < C_DIM; k0 += KT) {
    int cur = (k0 >> 5) & 1;
    WAIT_ASYNC();
    __syncthreads();
    if (k0 + KT < C_DIM) {
      int kn = k0 + KT, nxt = cur ^ 1;
      async_cp16(X + (long)(rBase + sr) * C_DIM + kn + sc,
                 &xs[nxt][sr * LDT + sc]);
      async_cp16(X + (long)(rBase + sr + 64) * C_DIM + kn + sc,
                 &xs[nxt][(sr + 64) * LDT + sc]);
      async_cp16(W + (long)(cBase + sr) * C_DIM + kn + sc,
                 &wt[nxt][sr * LDT + sc]);
    }
    const __bf16* xa = &xs[cur][(wm * 32) * LDT];
    const __bf16* wb = &wt[cur][(wn * 32) * LDT];
    v16bf A0 = load_A(xa, LDT, lane, 0);
    v16bf A1 = load_A(xa + 16 * LDT, LDT, lane, 0);
    v16bf B0 = load_B_nk(wb, LDT, lane, 0);
    v16bf B1 = load_B_nk(wb + 16 * LDT, LDT, lane, 0);
    acc[0][0] = wmma_bf16(A0, B0, acc[0][0]);
    acc[0][1] = wmma_bf16(A0, B1, acc[0][1]);
    acc[1][0] = wmma_bf16(A1, B0, acc[1][0]);
    acc[1][1] = wmma_bf16(A1, B1, acc[1][1]);
  }

  int half = lane >> 4, nn = lane & 15;
#pragma unroll
  for (int mi = 0; mi < 2; ++mi)
#pragma unroll
    for (int ni = 0; ni < 2; ++ni) {
      int col = cBase + wn * 32 + ni * 16 + nn;
      float bias = Pb[col];
#pragma unroll
      for (int vv = 0; vv < 8; ++vv) {
        int rr = rBase + wm * 32 + mi * 16 + vv + (half << 3);
        if (rr < ROWS_T) out[(long)rr * C_DIM + col] = acc[mi][ni][vv] + bias;
      }
    }
}

// ---------------------------------------------------------------------------
extern "C" void kernel_launch(void* const* d_in, const int* in_sizes, int n_in,
                              void* d_out, int out_size, void* d_ws, size_t ws_size,
                              hipStream_t stream) {
  const float* x      = (const float*)d_in[0];
  const float* qkv_w  = (const float*)d_in[1];
  const float* proj_w = (const float*)d_in[2];
  const float* proj_b = (const float*)d_in[3];
  const float* rpe_w  = (const float*)d_in[4];
  float* out = (float*)d_out;

  char* p = (char*)d_ws;
  auto take = [&](size_t bytes) -> char* {
    char* r = p;
    p += (bytes + 255) & ~(size_t)255;
    return r;
  };
  __bf16* xbf  = (__bf16*)take((size_t)RPAD * C_DIM * 2);
  __bf16* wqkv = (__bf16*)take((size_t)3 * C_DIM * C_DIM * 2);
  __bf16* wprj = (__bf16*)take((size_t)C_DIM * C_DIM * 2);
  __bf16* qbuf = (__bf16*)take((size_t)BATCH * H_HEADS * NP * HD * 2);
  __bf16* kbuf = (__bf16*)take((size_t)BATCH * H_HEADS * NP * HD * 2);
  __bf16* vbuf = (__bf16*)take((size_t)BATCH * H_HEADS * NP * HD * 2);
  __bf16* obf  = (__bf16*)take((size_t)RPAD * C_DIM * 2);

  cvt_bf16<<<1024, 256, 0, stream>>>(x, xbf, (long)ROWS_T * C_DIM);
  cvt_bf16<<<1024, 256, 0, stream>>>(qkv_w, wqkv, (long)3 * C_DIM * C_DIM);
  cvt_bf16<<<512, 256, 0, stream>>>(proj_w, wprj, (long)C_DIM * C_DIM);
  // zero M-padding rows of the GEMM inputs and the q/k/v padding rows
  zero_bf16<<<64, 256, 0, stream>>>(xbf + (size_t)ROWS_T * C_DIM,
                                    (long)(RPAD - ROWS_T) * C_DIM);
  zero_bf16<<<64, 256, 0, stream>>>(obf + (size_t)ROWS_T * C_DIM,
                                    (long)(RPAD - ROWS_T) * C_DIM);
  zero_bf16<<<1024, 256, 0, stream>>>(qbuf,
                                      (long)3 * BATCH * H_HEADS * NP * HD);

  qkv_gemm<<<dim3(RPAD / MT, 3 * C_DIM / NT), 256, 0, stream>>>(
      xbf, wqkv, qbuf, kbuf, vbuf);
  flash_rpe<<<dim3(BATCH * H_HEADS, NP / 128), 256, 0, stream>>>(
      qbuf, kbuf, vbuf, rpe_w, obf);
  proj_gemm<<<dim3(RPAD / MT, C_DIM / NT), 256, 0, stream>>>(
      obf, wprj, proj_b, out);
}
